// SimpleGNN_14972255994227
// MI455X (gfx1250) — matrix operand order repaired and checked by reading, not compiled, not run on hardware
//
#include <hip/hip_runtime.h>

#define N_NODES 50000
#define N_EDGES 400000
#define DIM 128

// ---------- types ----------
typedef __attribute__((ext_vector_type(16))) __bf16          v16bf;
typedef __attribute__((ext_vector_type(8)))  float           v8f;
typedef __attribute__((ext_vector_type(16))) unsigned short  v16u;
typedef __attribute__((ext_vector_type(8)))  unsigned short  v8u;
typedef __attribute__((ext_vector_type(4)))  float           v4f;

// ---------- bf16 helpers (RNE) ----------
__device__ __forceinline__ unsigned short f2bf(float f) {
    union { float f; unsigned u; } c; c.f = f;
    unsigned u = c.u;
    unsigned r = u + 0x7FFFu + ((u >> 16) & 1u);
    return (unsigned short)(r >> 16);
}
__device__ __forceinline__ float bf2f(unsigned short s) {
    union { unsigned u; float f; } c; c.u = ((unsigned)s) << 16;
    return c.f;
}

// ---------- zero agg + deg ----------
__global__ __launch_bounds__(256) void gnn_zero_kernel(float* __restrict__ agg,
                                                       float* __restrict__ deg) {
    int i = blockIdx.x * 256 + threadIdx.x;
    if (i < N_NODES * DIM) agg[i] = 0.0f;
    if (i < N_NODES)       deg[i] = 0.0f;
}

// ---------- WMMA GEMM:  Out[m,n] = act( A[m,:] . W[n,:] + bias[n] ) ----------
// One wave computes a 16x128 output tile (8 accumulators of 16x16).
// Block = 256 threads (8 waves) -> 128 rows per block.
// W (fp32 [DIM][DIM]) is converted to bf16 into LDS once per block.
template <bool RELU, bool IN_F32, bool OUT_BF16>
__global__ __launch_bounds__(256) void gnn_gemm_kernel(const void*  __restrict__ Ain,
                                                       const float* __restrict__ W,
                                                       const float* __restrict__ bias,
                                                       void*        __restrict__ Out) {
    __shared__ __align__(32) unsigned short Wlds[DIM * DIM];
    for (int i = threadIdx.x; i < DIM * DIM; i += 256) Wlds[i] = f2bf(W[i]);
    __syncthreads();

    const int lane = threadIdx.x & 31;
    const int wave = threadIdx.x >> 5;
    const int row0 = blockIdx.x * 128 + wave * 16;
    const int mloc = lane & 15;     // row within A-frag / col within B,C,D frags
    const int hi   = lane >> 4;     // which half of the wave

    int mrow = row0 + mloc;
    if (mrow >= N_NODES) mrow = N_NODES - 1;  // clamp loads; stores are guarded

    v8f acc[8] = {};   // 8 tiles of 16x16 f32

    #pragma unroll
    for (int kc = 0; kc < 4; ++kc) {
        const int kb = kc * 32;

        // ---- A fragment: 16x32 bf16 (two contiguous 8-element runs per lane) ----
        v16u a_u;
        if (IN_F32) {
            const float* ap = (const float*)Ain + (size_t)mrow * DIM + kb + hi * 8;
            v4f f0 = *(const v4f*)(ap + 0);
            v4f f1 = *(const v4f*)(ap + 4);
            v4f f2 = *(const v4f*)(ap + 16);
            v4f f3 = *(const v4f*)(ap + 20);
            #pragma unroll
            for (int i = 0; i < 4; ++i) {
                a_u[i]      = f2bf(f0[i]);
                a_u[4 + i]  = f2bf(f1[i]);
                a_u[8 + i]  = f2bf(f2[i]);
                a_u[12 + i] = f2bf(f3[i]);
            }
        } else {
            const unsigned short* ap =
                (const unsigned short*)Ain + (size_t)mrow * DIM + kb + hi * 8;
            v8u lo = *(const v8u*)(ap);
            v8u hh = *(const v8u*)(ap + 16);
            #pragma unroll
            for (int i = 0; i < 8; ++i) { a_u[i] = lo[i]; a_u[8 + i] = hh[i]; }
        }
        v16bf a = __builtin_bit_cast(v16bf, a_u);

        // ---- 8 N-tiles: B fragment from LDS, then WMMA ----
        #pragma unroll
        for (int nt = 0; nt < 8; ++nt) {
            const int n = nt * 16 + mloc;
            const v8u blo = *(const v8u*)&Wlds[n * DIM + kb + hi * 8];
            const v8u bhi = *(const v8u*)&Wlds[n * DIM + kb + hi * 8 + 16];
            v16u bu;
            #pragma unroll
            for (int i = 0; i < 8; ++i) { bu[i] = blo[i]; bu[8 + i] = bhi[i]; }
            v16bf b = __builtin_bit_cast(v16bf, bu);
            acc[nt] = __builtin_amdgcn_wmma_f32_16x16x32_bf16(
                false, a, false, b, (short)0, acc[nt], false, false);
        }
    }

    // ---- epilogue: bias (+ReLU), store ----
    #pragma unroll
    for (int nt = 0; nt < 8; ++nt) {
        const int n  = nt * 16 + mloc;
        const float bv = bias[n];
        #pragma unroll
        for (int j = 0; j < 8; ++j) {
            const int m = row0 + j + hi * 8;
            if (m < N_NODES) {
                float v = acc[nt][j] + bv;
                if (RELU) v = v > 0.0f ? v : 0.0f;
                if (OUT_BF16)
                    ((unsigned short*)Out)[(size_t)m * DIM + n] = f2bf(v);
                else
                    ((float*)Out)[(size_t)m * DIM + n] = v;
            }
        }
    }
}

// ---------- edge scatter: agg[dst] += h[src]; deg[dst] += 1 ----------
// one wave per edge; each lane handles 4 channels (L2-resident h, L2 atomics)
__global__ __launch_bounds__(256) void gnn_scatter_kernel(const long long* __restrict__ ei,
                                                          const unsigned short* __restrict__ h,
                                                          float* __restrict__ agg,
                                                          float* __restrict__ deg) {
    const int tid  = blockIdx.x * 256 + threadIdx.x;
    const int e    = tid >> 5;
    if (e >= N_EDGES) return;
    const int lane = tid & 31;
    const int src  = (int)ei[e];
    const int dst  = (int)ei[N_EDGES + e];
    const int c0   = lane * 4;

    const ushort4 hv = *(const ushort4*)(h + (size_t)src * DIM + c0);
    float* ap = agg + (size_t)dst * DIM + c0;
    atomicAdd(ap + 0, bf2f(hv.x));
    atomicAdd(ap + 1, bf2f(hv.y));
    atomicAdd(ap + 2, bf2f(hv.z));
    atomicAdd(ap + 3, bf2f(hv.w));
    if (lane == 0) atomicAdd(deg + dst, 1.0f);
}

// ---------- finalize: z = bf16( agg/max(deg,1) + h ) ----------
__global__ __launch_bounds__(256) void gnn_finalize_kernel(const float* __restrict__ agg,
                                                           const float* __restrict__ deg,
                                                           const unsigned short* __restrict__ h,
                                                           unsigned short* __restrict__ z) {
    int i = blockIdx.x * 256 + threadIdx.x;
    if (i >= N_NODES * DIM) return;
    const int node = i >> 7;
    float d = deg[node];
    d = d < 1.0f ? 1.0f : d;
    z[i] = f2bf(agg[i] / d + bf2f(h[i]));
}

// ---------- launcher ----------
extern "C" void kernel_launch(void* const* d_in, const int* in_sizes, int n_in,
                              void* d_out, int out_size, void* d_ws, size_t ws_size,
                              hipStream_t stream) {
    (void)in_sizes; (void)n_in; (void)out_size; (void)ws_size;

    const float*     x  = (const float*)d_in[0];
    const long long* ei = (const long long*)d_in[1];   // int64 edge_index [2][E]
    const float*     W1 = (const float*)d_in[2];
    const float*     b1 = (const float*)d_in[3];
    const float*     W2 = (const float*)d_in[4];
    const float*     b2 = (const float*)d_in[5];
    float*           out = (float*)d_out;

    // workspace layout (256B aligned)
    char* ws = (char*)d_ws;
    const size_t OFF_H   = 0;                       // bf16 h   : 12.8 MB
    const size_t OFF_Z   = 12800000;                // bf16 z   : 12.8 MB
    const size_t OFF_AGG = 25600000;                // f32 agg  : 25.6 MB
    const size_t OFF_DEG = 51200000;                // f32 deg  : 0.2 MB
    unsigned short* h   = (unsigned short*)(ws + OFF_H);
    unsigned short* z   = (unsigned short*)(ws + OFF_Z);
    float*          agg = (float*)(ws + OFF_AGG);
    float*          deg = (float*)(ws + OFF_DEG);

    const int gemm_blocks = (N_NODES + 127) / 128;           // 391
    const int elem_blocks = (N_NODES * DIM + 255) / 256;     // 25000
    const int edge_blocks = (N_EDGES * 32 + 255) / 256;      // 50000

    gnn_zero_kernel<<<elem_blocks, 256, 0, stream>>>(agg, deg);

    // fc1: h = relu(x @ W1^T + b1)   (fp32 in, bf16 out)
    gnn_gemm_kernel<true, true, true><<<gemm_blocks, 256, 0, stream>>>(
        (const void*)x, W1, b1, (void*)h);

    // scatter-add over edges
    gnn_scatter_kernel<<<edge_blocks, 256, 0, stream>>>(ei, h, agg, deg);

    // z = agg/deg + h
    gnn_finalize_kernel<<<elem_blocks, 256, 0, stream>>>(agg, deg, h, z);

    // fc2: out = z @ W2^T + b2       (bf16 in, fp32 out)
    gnn_gemm_kernel<false, false, false><<<gemm_blocks, 256, 0, stream>>>(
        (const void*)z, W2, b2, (void*)out);
}
